// GCN_36069135352114
// MI455X (gfx1250) — matrix operand order repaired and checked by reading, not compiled, not run on hardware
//
#include <hip/hip_runtime.h>
#include <math.h>

#define N_NODES  50000
#define N_EDGES  800000
#define ET       (N_EDGES + N_NODES)   // edges + self loops
#define IN_C     128
#define HID      16
#define HEADS    8
#define HH       (HEADS * HID)         // 128
#define OUT_C    16
#define NEG_SLOPE 0.2f

typedef __bf16 v16bf __attribute__((ext_vector_type(16)));
typedef float  v8f   __attribute__((ext_vector_type(8)));

union AV { uint4 q[2]; v16bf b; };

__device__ __forceinline__ unsigned pack2bf(float lo, float hi) {
    union { __bf16 h[2]; unsigned u; } v;
    v.h[0] = (__bf16)lo;
    v.h[1] = (__bf16)hi;
    return v.u;
}

// order-preserving float<->uint for atomicMax-based segment max
__device__ __forceinline__ unsigned encf(float f) {
    unsigned u = __float_as_uint(f);
    return (u & 0x80000000u) ? ~u : (u | 0x80000000u);
}
__device__ __forceinline__ float decf(unsigned e) {
    return (e & 0x80000000u) ? __uint_as_float(e ^ 0x80000000u)
                             : __uint_as_float(~e);
}

__device__ __forceinline__ float lrelu(float x) {
    return x > 0.f ? x : x * NEG_SLOPE;
}

__device__ __forceinline__ void edge_sd(const int* __restrict__ ei, int e,
                                        int& s, int& d) {
    if (e < N_EDGES) { s = ei[e]; d = ei[N_EDGES + e]; }
    else             { s = e - N_EDGES; d = s; }
}

// ---------------- conversion / prepack kernels ------------------------------
// x (fp32 row-major) -> packed bf16 row-major (pairs per dword)
__global__ __launch_bounds__(256) void k_cvt_x(const float* __restrict__ in,
                                               unsigned* __restrict__ outp,
                                               int npairs) {
    int t = blockIdx.x * blockDim.x + threadIdx.x;
    if (t >= npairs) return;
    outp[t] = pack2bf(in[2 * t], in[2 * t + 1]);
}

// W1 [128,128] -> B-fragment layout [ct(8)][ks(4)][lane(32)][8 dwords]
__global__ __launch_bounds__(256) void k_prepack_w1(const float* __restrict__ W,
                                                    unsigned* __restrict__ wpk) {
    int t = blockIdx.x * blockDim.x + threadIdx.x;
    if (t >= 8 * 4 * 32 * 8) return;
    int j  = t & 7;
    int l  = (t >> 3) & 31;
    int ks = (t >> 8) & 3;
    int ct = t >> 10;
    int K   = ks * 32 + ((j >= 4) ? 16 : 0) + ((l >= 16) ? 8 : 0) + 2 * (j & 3);
    int col = ct * 16 + (l & 15);
    wpk[t] = pack2bf(W[(size_t)K * HH + col], W[(size_t)(K + 1) * HH + col]);
}

// W2 [128,16] -> B-fragment layout [ks(4)][lane(32)][8 dwords]
__global__ __launch_bounds__(256) void k_prepack_w2(const float* __restrict__ W,
                                                    unsigned* __restrict__ wpk) {
    int t = blockIdx.x * blockDim.x + threadIdx.x;
    if (t >= 4 * 32 * 8) return;
    int j  = t & 7;
    int l  = (t >> 3) & 31;
    int ks = t >> 8;
    int K   = ks * 32 + ((j >= 4) ? 16 : 0) + ((l >= 16) ? 8 : 0) + 2 * (j & 3);
    int col = l & 15;
    wpk[t] = pack2bf(W[(size_t)K * OUT_C + col], W[(size_t)(K + 1) * OUT_C + col]);
}

// ---------------- GEMM1: h1 = x @ W1  (bf16 WMMA, fp32 accum) ---------------
// A fragment per kstep: lanes 0-15 need K {0..7,16..23} = row dwords {0-3,8-11},
// lanes 16-31 need K {8..15,24..31} = dwords {4-7,12-15} -> two uint4 loads.
__global__ __launch_bounds__(256) void k_gemm1(const uint4* __restrict__ xbf,
                                               const uint4* __restrict__ wpk,
                                               float* __restrict__ h) {
    const int lane   = threadIdx.x & 31;
    const int wave   = blockIdx.x * (blockDim.x >> 5) + (threadIdx.x >> 5);
    const int row0   = wave * 16;
    if (row0 >= N_NODES) return;
    const int m      = lane & 15;
    const int hiHalf = lane >> 4;

    AV a[4];
    const uint4* xr = xbf + (size_t)(row0 + m) * 16;   // 16 uint4 per row
#pragma unroll
    for (int ks = 0; ks < 4; ++ks) {
        a[ks].q[0] = xr[ks * 4 + hiHalf];
        a[ks].q[1] = xr[ks * 4 + 2 + hiHalf];
    }
#pragma unroll
    for (int ct = 0; ct < 8; ++ct) {
        v8f c = {};
#pragma unroll
        for (int ks = 0; ks < 4; ++ks) {
            AV b;
            const uint4* bp = wpk + ((size_t)(ct * 4 + ks) * 32 + lane) * 2;
            b.q[0] = bp[0];
            b.q[1] = bp[1];
            c = __builtin_amdgcn_wmma_f32_16x16x32_bf16(
                    false, a[ks].b, false, b.b, (short)0, c, false, false);
        }
#pragma unroll
        for (int j = 0; j < 8; ++j)
            h[(size_t)(row0 + j + hiHalf * 8) * HH + ct * 16 + m] = c[j];
    }
}

// ---------------- GEMM2: h2 = h1post(bf16) @ W2 ------------------------------
__global__ __launch_bounds__(256) void k_gemm2(const uint4* __restrict__ hbf,
                                               const uint4* __restrict__ wpk,
                                               float* __restrict__ hout) {
    const int lane   = threadIdx.x & 31;
    const int wave   = blockIdx.x * (blockDim.x >> 5) + (threadIdx.x >> 5);
    const int row0   = wave * 16;
    if (row0 >= N_NODES) return;
    const int m      = lane & 15;
    const int hiHalf = lane >> 4;

    const uint4* xr = hbf + (size_t)(row0 + m) * 16;
    v8f c = {};
#pragma unroll
    for (int ks = 0; ks < 4; ++ks) {
        AV a, b;
        a.q[0] = xr[ks * 4 + hiHalf];
        a.q[1] = xr[ks * 4 + 2 + hiHalf];
        const uint4* bp = wpk + ((size_t)ks * 32 + lane) * 2;
        b.q[0] = bp[0];
        b.q[1] = bp[1];
        c = __builtin_amdgcn_wmma_f32_16x16x32_bf16(
                false, a.b, false, b.b, (short)0, c, false, false);
    }
#pragma unroll
    for (int j = 0; j < 8; ++j)
        hout[(size_t)(row0 + j + hiHalf * 8) * OUT_C + m] = c[j];
}

// ---------------- attention logits ------------------------------------------
__global__ __launch_bounds__(256) void k_attn1(const float* __restrict__ h,
                                               const float* __restrict__ as,
                                               const float* __restrict__ ad,
                                               float* __restrict__ alS,
                                               float* __restrict__ alD) {
    int t = blockIdx.x * blockDim.x + threadIdx.x;
    if (t >= N_NODES * HEADS) return;
    int n = t >> 3, hd = t & 7;
    const float* hp = h + (size_t)n * HH + hd * HID;
    float s = 0.f, d = 0.f;
#pragma unroll
    for (int c = 0; c < HID; ++c) {
        float v = hp[c];
        s += v * as[hd * HID + c];
        d += v * ad[hd * HID + c];
    }
    alS[t] = s; alD[t] = d;
}

__global__ __launch_bounds__(256) void k_attn2(const float* __restrict__ h,
                                               const float* __restrict__ as,
                                               const float* __restrict__ ad,
                                               float* __restrict__ alS,
                                               float* __restrict__ alD) {
    int n = blockIdx.x * blockDim.x + threadIdx.x;
    if (n >= N_NODES) return;
    const float* hp = h + (size_t)n * OUT_C;
    float s = 0.f, d = 0.f;
#pragma unroll
    for (int c = 0; c < OUT_C; ++c) {
        float v = hp[c];
        s += v * as[c];
        d += v * ad[c];
    }
    alS[n] = s; alD[n] = d;
}

// ---------------- init kernels ----------------------------------------------
__global__ __launch_bounds__(256) void k_init1(unsigned* __restrict__ m1,
                                               float* __restrict__ den1,
                                               float* __restrict__ out1) {
    int t = blockIdx.x * blockDim.x + threadIdx.x;
    if (t < N_NODES * IN_C) out1[t] = 0.f;
    if (t < N_NODES * HEADS) { m1[t] = encf(-3.0e38f); den1[t] = 0.f; }
}

__global__ __launch_bounds__(256) void k_init2(unsigned* __restrict__ m2,
                                               float* __restrict__ den2,
                                               const float* __restrict__ b2,
                                               float* __restrict__ out) {
    int t = blockIdx.x * blockDim.x + threadIdx.x;
    if (t < N_NODES * OUT_C) out[t] = b2[t & 15];
    if (t < N_NODES) { m2[t] = encf(-3.0e38f); den2[t] = 0.f; }
}

// ---------------- layer-1 edge phase ----------------------------------------
__global__ __launch_bounds__(256) void k_edge_max1(const int* __restrict__ ei,
                                                   const float* __restrict__ alS,
                                                   const float* __restrict__ alD,
                                                   unsigned* __restrict__ m1) {
    int e = blockIdx.x * blockDim.x + threadIdx.x;
    if (e >= ET) return;
    int s, d; edge_sd(ei, e, s, d);
#pragma unroll
    for (int hd = 0; hd < HEADS; ++hd) {
        float sc = lrelu(alS[s * HEADS + hd] + alD[d * HEADS + hd]);
        atomicMax(&m1[d * HEADS + hd], encf(sc));
    }
}

__global__ __launch_bounds__(256) void k_edge_sum1(const int* __restrict__ ei,
                                                   const float* __restrict__ alS,
                                                   const float* __restrict__ alD,
                                                   const unsigned* __restrict__ m1,
                                                   float* __restrict__ den1) {
    int e = blockIdx.x * blockDim.x + threadIdx.x;
    if (e >= ET) return;
    int s, d; edge_sd(ei, e, s, d);
#pragma unroll
    for (int hd = 0; hd < HEADS; ++hd) {
        float sc = lrelu(alS[s * HEADS + hd] + alD[d * HEADS + hd]);
        float w  = __expf(sc - decf(m1[d * HEADS + hd]));
        atomicAdd(&den1[d * HEADS + hd], w);
    }
}

__global__ __launch_bounds__(256) void k_scatter1(const int* __restrict__ ei,
                                                  const float* __restrict__ alS,
                                                  const float* __restrict__ alD,
                                                  const unsigned* __restrict__ m1,
                                                  const float* __restrict__ den1,
                                                  const float* __restrict__ h,
                                                  float* __restrict__ out1) {
    int t = blockIdx.x * blockDim.x + threadIdx.x;
    if (t >= ET * HEADS) return;
    int e = t >> 3, hd = t & 7;
    int s, d; edge_sd(ei, e, s, d);
    float sc = lrelu(alS[s * HEADS + hd] + alD[d * HEADS + hd]);
    float alpha = __expf(sc - decf(m1[d * HEADS + hd])) / den1[d * HEADS + hd];
    const float* hp = h    + (size_t)s * HH + hd * HID;
    float*       op = out1 + (size_t)d * HH + hd * HID;
#pragma unroll
    for (int c = 0; c < HID; ++c)
        atomicAdd(&op[c], hp[c] * alpha);
}

// bias + ELU, fused with fp32 -> packed bf16 conversion (GEMM2 input)
__global__ __launch_bounds__(256) void k_bias_elu_pack(const float* __restrict__ out1,
                                                       const float* __restrict__ b1,
                                                       unsigned* __restrict__ hbf) {
    int t = blockIdx.x * blockDim.x + threadIdx.x;
    if (t >= N_NODES * (IN_C / 2)) return;
    int f = (2 * t) & 127;
    float v0 = out1[2 * t]     + b1[f];
    float v1 = out1[2 * t + 1] + b1[f + 1];
    v0 = v0 > 0.f ? v0 : (__expf(v0) - 1.f);
    v1 = v1 > 0.f ? v1 : (__expf(v1) - 1.f);
    hbf[t] = pack2bf(v0, v1);
}

// ---------------- layer-2 edge phase ----------------------------------------
__global__ __launch_bounds__(256) void k_edge_max2(const int* __restrict__ ei,
                                                   const float* __restrict__ alS,
                                                   const float* __restrict__ alD,
                                                   unsigned* __restrict__ m2) {
    int e = blockIdx.x * blockDim.x + threadIdx.x;
    if (e >= ET) return;
    int s, d; edge_sd(ei, e, s, d);
    float sc = lrelu(alS[s] + alD[d]);
    atomicMax(&m2[d], encf(sc));
}

__global__ __launch_bounds__(256) void k_edge_sum2(const int* __restrict__ ei,
                                                   const float* __restrict__ alS,
                                                   const float* __restrict__ alD,
                                                   const unsigned* __restrict__ m2,
                                                   float* __restrict__ den2) {
    int e = blockIdx.x * blockDim.x + threadIdx.x;
    if (e >= ET) return;
    int s, d; edge_sd(ei, e, s, d);
    float sc = lrelu(alS[s] + alD[d]);
    atomicAdd(&den2[d], __expf(sc - decf(m2[d])));
}

__global__ __launch_bounds__(256) void k_scatter2(const int* __restrict__ ei,
                                                  const float* __restrict__ alS,
                                                  const float* __restrict__ alD,
                                                  const unsigned* __restrict__ m2,
                                                  const float* __restrict__ den2,
                                                  const float* __restrict__ h,
                                                  float* __restrict__ out) {
    int e = blockIdx.x * blockDim.x + threadIdx.x;
    if (e >= ET) return;
    int s, d; edge_sd(ei, e, s, d);
    float sc = lrelu(alS[s] + alD[d]);
    float alpha = __expf(sc - decf(m2[d])) / den2[d];
    const float* hp = h   + (size_t)s * OUT_C;
    float*       op = out + (size_t)d * OUT_C;
#pragma unroll
    for (int c = 0; c < OUT_C; ++c)
        atomicAdd(&op[c], hp[c] * alpha);
}

// ---------------- launcher ---------------------------------------------------
extern "C" void kernel_launch(void* const* d_in, const int* in_sizes, int n_in,
                              void* d_out, int out_size, void* d_ws, size_t ws_size,
                              hipStream_t stream) {
    const float* x   = (const float*)d_in[0];
    const int*   ei  = (const int*)  d_in[1];   // [2, E] flat
    const float* W1  = (const float*)d_in[2];
    const float* aS1 = (const float*)d_in[3];
    const float* aD1 = (const float*)d_in[4];
    const float* b1  = (const float*)d_in[5];
    const float* W2  = (const float*)d_in[6];
    const float* aS2 = (const float*)d_in[7];
    const float* aD2 = (const float*)d_in[8];
    const float* b2  = (const float*)d_in[9];
    float* out = (float*)d_out;

    float* w = (float*)d_ws;
    float*    h1   = w;                                           // N*128 f32
    float*    out1 = h1   + (size_t)N_NODES * HH;                 // N*128 f32
    float*    alS1 = out1 + (size_t)N_NODES * HH;                 // N*8
    float*    alD1 = alS1 + (size_t)N_NODES * HEADS;              // N*8
    unsigned* m1   = (unsigned*)(alD1 + (size_t)N_NODES * HEADS); // N*8
    float*    den1 = (float*)(m1 + (size_t)N_NODES * HEADS);      // N*8
    float*    h2   = den1 + (size_t)N_NODES * HEADS;              // N*16
    float*    alS2 = h2   + (size_t)N_NODES * OUT_C;              // N
    float*    alD2 = alS2 + N_NODES;                              // N
    unsigned* m2   = (unsigned*)(alD2 + N_NODES);                 // N
    float*    den2 = (float*)(m2 + N_NODES);                      // N
    unsigned* xbf  = (unsigned*)(den2 + N_NODES);                 // N*64 dwords
    unsigned* hbf  = xbf + (size_t)N_NODES * (IN_C / 2);          // N*64 dwords
    unsigned* w1pk = hbf + (size_t)N_NODES * (IN_C / 2);          // 8192 dwords
    unsigned* w2pk = w1pk + 8192;                                 // 1024 dwords

    const int TB = 256;
    const int waves = N_NODES / 16;                 // 3125, exact
    const int gemmBlocks = (waves + 7) / 8;         // 8 waves/block
    const int edgeBlocks = (ET + TB - 1) / TB;
    const int pairBlocks = (N_NODES * (IN_C / 2) + TB - 1) / TB;

    // ---- prep ----
    k_cvt_x<<<pairBlocks, TB, 0, stream>>>(x, xbf, N_NODES * (IN_C / 2));
    k_prepack_w1<<<(8192 + TB - 1) / TB, TB, 0, stream>>>(W1, w1pk);
    k_prepack_w2<<<(1024 + TB - 1) / TB, TB, 0, stream>>>(W2, w2pk);

    // ---- layer 1 ----
    k_gemm1<<<gemmBlocks, TB, 0, stream>>>((const uint4*)xbf, (const uint4*)w1pk, h1);
    k_attn1<<<(N_NODES * HEADS + TB - 1) / TB, TB, 0, stream>>>(h1, aS1, aD1, alS1, alD1);
    k_init1<<<(N_NODES * IN_C + TB - 1) / TB, TB, 0, stream>>>(m1, den1, out1);
    k_edge_max1<<<edgeBlocks, TB, 0, stream>>>(ei, alS1, alD1, m1);
    k_edge_sum1<<<edgeBlocks, TB, 0, stream>>>(ei, alS1, alD1, m1, den1);
    k_scatter1<<<((size_t)ET * HEADS + TB - 1) / TB, TB, 0, stream>>>(
        ei, alS1, alD1, m1, den1, h1, out1);
    k_bias_elu_pack<<<pairBlocks, TB, 0, stream>>>(out1, b1, hbf);

    // ---- layer 2 ----
    k_gemm2<<<gemmBlocks, TB, 0, stream>>>((const uint4*)hbf, (const uint4*)w2pk, h2);
    k_attn2<<<(N_NODES + TB - 1) / TB, TB, 0, stream>>>(h2, aS2, aD2, alS2, alD2);
    k_init2<<<(N_NODES * OUT_C + TB - 1) / TB, TB, 0, stream>>>(m2, den2, b2, out);
    k_edge_max2<<<edgeBlocks, TB, 0, stream>>>(ei, alS2, alD2, m2);
    k_edge_sum2<<<edgeBlocks, TB, 0, stream>>>(ei, alS2, alD2, m2, den2);
    k_scatter2<<<edgeBlocks, TB, 0, stream>>>(ei, alS2, alD2, m2, den2, h2, out);
}
